// CrossAttention_13709535609715
// MI455X (gfx1250) — matrix operand order repaired and checked by reading, not compiled, not run on hardware
//
#include <hip/hip_runtime.h>
#include <hip/hip_bf16.h>

// CDNA5 (gfx1250) cross-attention + Sinkhorn.
//
// m = sinkhorn(sigmoid(Qh Kh^T)), y = (m Vh) Wo + bo
// Sinkhorn reduced to diagonal scaling: m = diag(u) X diag(v),
//   u_i = 1/(X v)_i , v_j = 1/(X^T u)_j  (5 alternations, v0 = 1)
// X is materialized once (fp32) into the m-region of d_out; the 10 Sinkhorn
// passes are read-only matvecs (L2-resident per-head 4MB tiles), then one
// in-place scale. All GEMMs use v_wmma_f32_16x16x32_f16.

typedef _Float16 half_t;
typedef __attribute__((ext_vector_type(16))) _Float16 v16h;
typedef __attribute__((ext_vector_type(8)))  _Float16 v8h;
typedef __attribute__((ext_vector_type(8)))  float    v8f;
typedef __attribute__((ext_vector_type(4)))  float    v4f;

#define HIDN 1024
#define SEQ  1024
#define NHD  16
#define DKD  64

// ---------------- WMMA fragment helpers (ISA 7.12.2 layouts) ----------------
// A (16x32 f16, MxK): lane l -> row = l&15; 16 halfs cover
//   K = h8+{0..7} and K = 16+h8+{0..7}, h8 = (l>>4)*8.
// B (32x16 f16, KxN) from a transposed [N][K] array: lane l -> col = l&15;
//   K = h8+{0..7} and K = 16+h8+{0..7}, h8 = (l>>4)*8 (same as A).
// C/D (16x16 f32): lane l, elem e -> row = e + (l>>4)*8, col = l&15.

__device__ __forceinline__ v16h load_a_frag_f32(const float* __restrict__ A, int lda,
                                                int row0, int kbase, int lane) {
  int r  = row0 + (lane & 15);
  int h8 = (lane >> 4) << 3;
  const float* p0 = A + (size_t)r * lda + kbase + h8;
  v4f a0 = *(const v4f*)(p0);
  v4f a1 = *(const v4f*)(p0 + 4);
  v4f b0 = *(const v4f*)(p0 + 16);
  v4f b1 = *(const v4f*)(p0 + 20);
  v16h f;
#pragma unroll
  for (int t = 0; t < 4; ++t) {
    f[t]      = (half_t)a0[t];
    f[4 + t]  = (half_t)a1[t];
    f[8 + t]  = (half_t)b0[t];
    f[12 + t] = (half_t)b1[t];
  }
  return f;
}

__device__ __forceinline__ v16h load_a_frag_f16(const half_t* __restrict__ A, int lda,
                                                int row0, int kbase, int lane) {
  int r  = row0 + (lane & 15);
  int h8 = (lane >> 4) << 3;
  const half_t* p0 = A + (size_t)r * lda + kbase + h8;
  v8h a = *(const v8h*)(p0);
  v8h b = *(const v8h*)(p0 + 16);
  v16h f;
#pragma unroll
  for (int t = 0; t < 8; ++t) { f[t] = a[t]; f[8 + t] = b[t]; }
  return f;
}

__device__ __forceinline__ v16h load_b_frag_f16(const half_t* __restrict__ BT, int ldb,
                                                int nbase, int kbase, int lane) {
  int c  = nbase + (lane & 15);
  int k0 = kbase + ((lane >> 4) << 3);
  const half_t* p = BT + (size_t)c * ldb + k0;
  v8h a = *(const v8h*)(p);
  v8h b = *(const v8h*)(p + 16);
  v16h f;
#pragma unroll
  for (int t = 0; t < 8; ++t) { f[t] = a[t]; f[8 + t] = b[t]; }
  return f;
}

__device__ __forceinline__ v8f wmma_f16(v16h a, v16h b, v8f c) {
  return __builtin_amdgcn_wmma_f32_16x16x32_f16(false, a, false, b, (short)0, c,
                                                false, false);
}

// ------------- K0: weight convert+transpose f32[K][N] -> f16[N][K] ----------
__global__ void wt_convert_kernel(const float* __restrict__ W0, const float* __restrict__ W1,
                                  const float* __restrict__ W2, const float* __restrict__ W3,
                                  half_t* __restrict__ dst) {
  const float* Ws[4] = {W0, W1, W2, W3};
  const float* src = Ws[blockIdx.z];
  half_t* d = dst + ((size_t)blockIdx.z << 20);
  __shared__ float tile[32][33];
  int x  = (blockIdx.x << 5) + threadIdx.x;   // column n of W
  int y0 = (blockIdx.y << 5) + threadIdx.y;   // row    k of W
#pragma unroll
  for (int j = 0; j < 32; j += 8)
    tile[threadIdx.y + j][threadIdx.x] = src[(size_t)(y0 + j) * HIDN + x];
  __syncthreads();
  int x2 = (blockIdx.y << 5) + threadIdx.x;   // k in WT
  int y2 = (blockIdx.x << 5) + threadIdx.y;   // n in WT
#pragma unroll
  for (int j = 0; j < 32; j += 8)
    d[(size_t)(y2 + j) * HIDN + x2] = (half_t)tile[threadIdx.x][threadIdx.y + j];
}

// ---- K1: input projection GEMM [4096x1024]x[1024x1024], f16 out, head layout
// transposed==0: dst[b][h][s][d]  (Qh/Kh);  transposed==1: dst[b][h][d][s] (V^T)
__global__ void proj_kernel(const float* __restrict__ X, const half_t* __restrict__ WT,
                            const float* __restrict__ bias, half_t* __restrict__ dst,
                            float scale, int transposed) {
  int lane = threadIdx.x & 31;
  int tile = blockIdx.x * 8 + (threadIdx.x >> 5);   // 0..4095
  int row0 = (tile >> 4) << 4;
  int n0   = (tile & 15) << 6;
  v8f acc[4] = {};
  for (int k = 0; k < HIDN; k += 32) {
    v16h a = load_a_frag_f32(X, HIDN, row0, k, lane);
#pragma unroll
    for (int j = 0; j < 4; ++j) {
      v16h b = load_b_frag_f16(WT, HIDN, n0 + (j << 4), k, lane);
      acc[j] = wmma_f16(a, b, acc[j]);
    }
  }
  int cn = lane & 15, rb = (lane >> 4) << 3;
#pragma unroll
  for (int j = 0; j < 4; ++j) {
#pragma unroll
    for (int e = 0; e < 8; ++e) {
      int n    = n0 + (j << 4) + cn;
      int mrow = row0 + rb + e;
      float val = (acc[j][e] + bias[n]) * scale;
      int bidx = mrow >> 10, s = mrow & (SEQ - 1);
      int h = n >> 6, d = n & (DKD - 1);
      if (!transposed)
        dst[((((size_t)bidx * NHD + h) << 10) + s) * DKD + d] = (half_t)val;
      else
        dst[((((size_t)bidx * NHD + h) * DKD + d) << 10) + s] = (half_t)val;
    }
  }
}

// ---------- K2: X = sigmoid(Qh Kh^T) per (b,h), fp32 into d_out m-region ----
__global__ void scores_kernel(const half_t* __restrict__ QH, const half_t* __restrict__ KH,
                              float* __restrict__ M) {
  int lane = threadIdx.x & 31;
  int tile = blockIdx.x * 8 + (threadIdx.x >> 5);   // 0..65535
  int bh = tile >> 10;
  int t  = tile & 1023;
  int row0 = (t >> 4) << 4;
  int n0   = (t & 15) << 6;
  const half_t* Q = QH + ((size_t)bh << 16);
  const half_t* K = KH + ((size_t)bh << 16);
  v8f acc[4] = {};
  for (int k = 0; k < DKD; k += 32) {
    v16h a = load_a_frag_f16(Q, DKD, row0, k, lane);
#pragma unroll
    for (int j = 0; j < 4; ++j) {
      v16h b = load_b_frag_f16(K, DKD, n0 + (j << 4), k, lane);
      acc[j] = wmma_f16(a, b, acc[j]);
    }
  }
  float* out = M + ((size_t)bh << 20);
  int cn = lane & 15, rb = (lane >> 4) << 3;
#pragma unroll
  for (int j = 0; j < 4; ++j)
#pragma unroll
    for (int e = 0; e < 8; ++e) {
      float x = acc[j][e];
      x = 1.0f / (1.0f + __expf(-x));
      out[((size_t)(row0 + rb + e) << 10) + n0 + (j << 4) + cn] = x;
    }
}

// ------------------------- Sinkhorn matvec passes ---------------------------
__global__ void init_ones_kernel(float* __restrict__ p) {
  p[blockIdx.x * 256 + threadIdx.x] = 1.0f;
}

// u_i = 1 / sum_j X_ij * v_j   (one wave per row)
__global__ void row_pass_kernel(const float* __restrict__ M, const float* __restrict__ vv,
                                float* __restrict__ uu) {
  int lane = threadIdx.x & 31;
  int row  = blockIdx.x * 8 + (threadIdx.x >> 5);   // 0..65535
  int bh = row >> 10, i = row & 1023;
  const float* Xr = M + (((size_t)bh << 10) + i) * SEQ;
  const float* vb = vv + (bh << 10);
  float s = 0.f;
#pragma unroll
  for (int t = 0; t < 8; ++t) {
    int j = (lane << 2) + (t << 7);
    v4f x = *(const v4f*)(Xr + j);
    v4f w = *(const v4f*)(vb + j);
    s += x[0] * w[0] + x[1] * w[1] + x[2] * w[2] + x[3] * w[3];
  }
#pragma unroll
  for (int off = 16; off >= 1; off >>= 1) s += __shfl_xor(s, off, 32);
  if (lane == 0) uu[row] = 1.0f / s;
}

// v_j = 1 / sum_i u_i * X_ij   (64 cols x 4 row-segments per block, LDS reduce)
__global__ void col_pass_kernel(const float* __restrict__ M, const float* __restrict__ uu,
                                float* __restrict__ vv) {
  __shared__ float part[4][64];
  int c = threadIdx.x & 63, seg = threadIdx.x >> 6;
  int bh = blockIdx.x >> 4;
  int j  = ((blockIdx.x & 15) << 6) + c;
  const float* Xb = M + ((size_t)bh << 20);
  const float* ub = uu + (bh << 10);
  float s = 0.f;
  int i0 = seg << 8;
  for (int i = 0; i < 256; ++i) {
    int r = i0 + i;
    s += ub[r] * Xb[((size_t)r << 10) + j];
  }
  part[seg][c] = s;
  __syncthreads();
  if (seg == 0) {
    float t = part[0][c] + part[1][c] + part[2][c] + part[3][c];
    vv[(bh << 10) + j] = 1.0f / t;
  }
}

// ------------- in-place: m_ij = X_ij * u_i * v_j  (float4 per thread) -------
__global__ void scale_kernel(float* __restrict__ M, const float* __restrict__ uu,
                             const float* __restrict__ vv) {
  size_t g = (size_t)blockIdx.x * 256 + threadIdx.x;  // 0..16M-1 float4s
  int j4  = (int)(g & 255);
  int row = (int)((g >> 8) & 1023);
  int bh  = (int)(g >> 18);
  v4f m4 = *(v4f*)(M + (g << 2));
  float ui = uu[(bh << 10) + row];
  v4f v4 = *(const v4f*)(vv + (bh << 10) + (j4 << 2));
  m4 *= ui;
  m4 *= v4;
  *(v4f*)(M + (g << 2)) = m4;
}

// -------------- y1 = m @ Vh  per (b,h): fp32 m tiles -> f16 WMMA ------------
__global__ void attn_out_kernel(const float* __restrict__ M, const half_t* __restrict__ VHT,
                                half_t* __restrict__ Y1) {
  int lane = threadIdx.x & 31;
  int tile = blockIdx.x * 8 + (threadIdx.x >> 5);   // 0..4095
  int bh   = tile >> 6;
  int row0 = (tile & 63) << 4;
  const float*  Mb = M   + ((size_t)bh << 20);
  const half_t* Vt = VHT + ((size_t)bh << 16);      // [DK][S]
  v8f acc[4] = {};
  for (int k = 0; k < SEQ; k += 32) {
    v16h a = load_a_frag_f32(Mb, SEQ, row0, k, lane);
#pragma unroll
    for (int j = 0; j < 4; ++j) {
      v16h b = load_b_frag_f16(Vt, SEQ, (j << 4), k, lane);
      acc[j] = wmma_f16(a, b, acc[j]);
    }
  }
  int b_ = bh >> 4, h = bh & 15;
  int cn = lane & 15, rb = (lane >> 4) << 3;
#pragma unroll
  for (int j = 0; j < 4; ++j)
#pragma unroll
    for (int e = 0; e < 8; ++e) {
      int s = row0 + rb + e;
      int d = (j << 4) + cn;
      Y1[(((size_t)b_ << 10) + s) * HIDN + (h << 6) + d] = (half_t)acc[j][e];
    }
}

// ------------------ y = Y1 @ Wo + bo -> fp32 d_out y-region -----------------
__global__ void out_proj_kernel(const half_t* __restrict__ Y1, const half_t* __restrict__ WoT,
                                const float* __restrict__ bo, float* __restrict__ Y) {
  int lane = threadIdx.x & 31;
  int tile = blockIdx.x * 8 + (threadIdx.x >> 5);
  int row0 = (tile >> 4) << 4;
  int n0   = (tile & 15) << 6;
  v8f acc[4] = {};
  for (int k = 0; k < HIDN; k += 32) {
    v16h a = load_a_frag_f16(Y1, HIDN, row0, k, lane);
#pragma unroll
    for (int j = 0; j < 4; ++j) {
      v16h b = load_b_frag_f16(WoT, HIDN, n0 + (j << 4), k, lane);
      acc[j] = wmma_f16(a, b, acc[j]);
    }
  }
  int cn = lane & 15, rb = (lane >> 4) << 3;
#pragma unroll
  for (int j = 0; j < 4; ++j)
#pragma unroll
    for (int e = 0; e < 8; ++e) {
      int n    = n0 + (j << 4) + cn;
      int mrow = row0 + rb + e;
      Y[(size_t)mrow * HIDN + n] = acc[j][e] + bo[n];
    }
}

// ----------------------------------------------------------------------------
extern "C" void kernel_launch(void* const* d_in, const int* in_sizes, int n_in,
                              void* d_out, int out_size, void* d_ws, size_t ws_size,
                              hipStream_t stream) {
  (void)in_sizes; (void)n_in; (void)out_size; (void)ws_size;
  const float* q  = (const float*)d_in[0];
  const float* k  = (const float*)d_in[1];
  const float* v  = (const float*)d_in[2];
  const float* Wq = (const float*)d_in[3];
  const float* bq = (const float*)d_in[4];
  const float* Wk = (const float*)d_in[5];
  const float* bk = (const float*)d_in[6];
  const float* Wv = (const float*)d_in[7];
  const float* bv = (const float*)d_in[8];
  const float* Wo = (const float*)d_in[9];
  const float* bo = (const float*)d_in[10];

  float* M = (float*)d_out;                       // m: 4*16*1024*1024 fp32
  float* Y = M + (size_t)67108864;                // y: 4*1024*1024 fp32

  // workspace layout (~40.5 MB)
  char*   ws   = (char*)d_ws;
  half_t* WT   = (half_t*)ws;                     // 4x [1024][1024] f16 (transposed)
  half_t* QH   = (half_t*)(ws + ((size_t)8  << 20));  // [B][H][S][DK] f16
  half_t* KH   = (half_t*)(ws + ((size_t)16 << 20));  // [B][H][S][DK] f16
  half_t* VHT  = (half_t*)(ws + ((size_t)24 << 20));  // [B][H][DK][S] f16
  half_t* Y1   = (half_t*)(ws + ((size_t)32 << 20));  // [B][S][HID]   f16
  float*  uvec = (float*) (ws + ((size_t)40 << 20));  // [B*H*S] fp32
  float*  vvec = uvec + 65536;                        // [B*H*S] fp32

  const float scale = 1.0f / 32.0f;               // HID^-0.5

  wt_convert_kernel<<<dim3(32, 32, 4), dim3(32, 8), 0, stream>>>(Wq, Wk, Wv, Wo, WT);

  proj_kernel<<<512, 256, 0, stream>>>(q, WT,                 bq, QH,  scale, 0);
  proj_kernel<<<512, 256, 0, stream>>>(k, WT + (1u << 20),    bk, KH,  1.0f,  0);
  proj_kernel<<<512, 256, 0, stream>>>(v, WT + (2u << 20),    bv, VHT, 1.0f,  1);

  scores_kernel<<<8192, 256, 0, stream>>>(QH, KH, M);

  init_ones_kernel<<<256, 256, 0, stream>>>(vvec);
  for (int it = 0; it < 5; ++it) {
    row_pass_kernel<<<8192, 256, 0, stream>>>(M, vvec, uvec);
    col_pass_kernel<<<1024, 256, 0, stream>>>(M, uvec, vvec);
  }
  scale_kernel<<<65536, 256, 0, stream>>>(M, uvec, vvec);

  attn_out_kernel<<<512, 256, 0, stream>>>(M, VHT, Y1);
  out_proj_kernel<<<512, 256, 0, stream>>>(Y1, WT + (3u << 20), bo, Y);
}
